// Model_2147483648341
// MI455X (gfx1250) — matrix operand order repaired and checked by reading, not compiled, not run on hardware
//
#include <hip/hip_runtime.h>
#include <hip/hip_bf16.h>

typedef __bf16 bf16;
typedef __attribute__((ext_vector_type(16))) __bf16 v16bf;
typedef __attribute__((ext_vector_type(8)))  __bf16 v8bf;
typedef __attribute__((ext_vector_type(8)))  float  v8f;

#define B_    32
#define L_    512
#define NV    256
#define DMODEL 512
#define DSTATE 16
#define DFF   512
#define PLEN  96
#define DTR   32
#define NMARK 6
#define LM    262            // NV + NMARK tokens
#define MTOK  (B_*LM)        // 8384 rows (divisible by 32)

// ---------------------------------------------------------------------------
// WMMA fragment loaders (contiguous k = 16*half + i order, same for A and B, wave32)
// A 16x32 (MxK): lanes 0-15 -> M=lane, elems i -> K=i
//                lanes16-31 -> M=lane-16, elems i -> K=16+i
// B 32x16 (KxN): lane n=lane&15; half(lane>=16): elems i -> K = half*16 + i
// ---------------------------------------------------------------------------
__device__ inline v16bf load_a_frag(const bf16* __restrict__ A, int lda) {
  int lane = threadIdx.x & 31;
  int half = lane >> 4, mr = lane & 15;
  const bf16* row = A + mr * lda + half * 16;
  v8bf lo = *(const v8bf*)(row);
  v8bf hi = *(const v8bf*)(row + 8);
  v16bf f;
#pragma unroll
  for (int i = 0; i < 8; ++i) { f[i] = lo[i]; f[i + 8] = hi[i]; }
  return f;
}

__device__ inline v16bf load_b_frag(const bf16* __restrict__ W, int ldw) {
  int lane = threadIdx.x & 31;
  int half = lane >> 4, nr = lane & 15;
  return *(const v16bf*)(W + nr * ldw + half * 16);
}

// C = act(A[M,K](bf16) @ W[N,K]^T(bf16) + bias) (+C if ACC); optional bf16 mirror.
// One wave per block computes a (16*MT) x (16*NT) tile. The k-loop is manually
// unrolled x2 with two statically-named fragment buffer sets so the double
// buffering stays in fixed VGPRs (no v_movrels / scratch spills).
template <int MT, int NT, int ACT, int ACC, int WBF>
__global__ __launch_bounds__(32)
void gemm_wmma_kernel(const bf16* __restrict__ A, const bf16* __restrict__ W,
                      const float* __restrict__ bias,
                      float* __restrict__ C, bf16* __restrict__ Cbf,
                      int M, int N, int K) {
  const int m0 = blockIdx.x * 16 * MT;
  const int n0 = blockIdx.y * 16 * NT;
  const int lane = threadIdx.x & 31;
  const int half = lane >> 4, lr = lane & 15;

  v8f acc[MT][NT];
#pragma unroll
  for (int i = 0; i < MT; ++i)
#pragma unroll
    for (int t = 0; t < NT; ++t)
#pragma unroll
      for (int r = 0; r < 8; ++r) acc[i][t][r] = 0.0f;

  v16bf a0[MT], b0[NT], a1[MT], b1[NT];
#pragma unroll
  for (int i = 0; i < MT; ++i) a0[i] = load_a_frag(A + (m0 + i * 16) * K, K);
#pragma unroll
  for (int t = 0; t < NT; ++t) b0[t] = load_b_frag(W + (n0 + t * 16) * K, K);

  int k0 = 0;
  for (;;) {
    const bool more1 = (k0 + 32) < K;
    if (more1) {
      __builtin_prefetch(A + (m0 + lr) * K + k0 + 64, 0, 3);
#pragma unroll
      for (int i = 0; i < MT; ++i)
        a1[i] = load_a_frag(A + (m0 + i * 16) * K + k0 + 32, K);
#pragma unroll
      for (int t = 0; t < NT; ++t)
        b1[t] = load_b_frag(W + (n0 + t * 16) * K + k0 + 32, K);
    }
#pragma unroll
    for (int i = 0; i < MT; ++i)
#pragma unroll
      for (int t = 0; t < NT; ++t)
        acc[i][t] = __builtin_amdgcn_wmma_f32_16x16x32_bf16(
            false, a0[i], false, b0[t], (short)0, acc[i][t], false, false);
    if (!more1) break;

    const bool more2 = (k0 + 64) < K;
    if (more2) {
#pragma unroll
      for (int i = 0; i < MT; ++i)
        a0[i] = load_a_frag(A + (m0 + i * 16) * K + k0 + 64, K);
#pragma unroll
      for (int t = 0; t < NT; ++t)
        b0[t] = load_b_frag(W + (n0 + t * 16) * K + k0 + 64, K);
    }
#pragma unroll
    for (int i = 0; i < MT; ++i)
#pragma unroll
      for (int t = 0; t < NT; ++t)
        acc[i][t] = __builtin_amdgcn_wmma_f32_16x16x32_bf16(
            false, a1[i], false, b1[t], (short)0, acc[i][t], false, false);
    if (!more2) break;
    k0 += 64;
  }

#pragma unroll
  for (int i = 0; i < MT; ++i)
#pragma unroll
    for (int t = 0; t < NT; ++t) {
      const int n = n0 + t * 16 + lr;
      const float bv = bias ? bias[n] : 0.0f;
#pragma unroll
      for (int r = 0; r < 8; ++r) {
        const int m = m0 + i * 16 + half * 8 + r;
        float v = acc[i][t][r] + bv;
        if constexpr (ACC) v += C[m * N + n];
        if constexpr (ACT == 1) v = v > 0.0f ? v : 0.0f;
        C[m * N + n] = v;
        if constexpr (WBF) Cbf[m * N + n] = (bf16)v;
      }
    }
}

// ---------------------------------------------------------------------------
// Elementwise / reduction kernels
// ---------------------------------------------------------------------------
__global__ void cvt_bf16_kernel(const float* __restrict__ s, bf16* __restrict__ d, int n) {
  int i = blockIdx.x * 256 + threadIdx.x;
  if (i < n) d[i] = (bf16)s[i];
}

// masked per-(b,n) statistics over L
__global__ void stats_kernel(const float* __restrict__ x,
                             float* __restrict__ means, float* __restrict__ stdev) {
  const int b = blockIdx.x;
  const int n = blockIdx.y * 32 + (threadIdx.x & 31);
  const int slice = threadIdx.x >> 5;  // 0..7
  float sv = 0.f, sx = 0.f, sxx = 0.f;
  for (int i = 0; i < L_ / 8; ++i) {
    const int l = slice * (L_ / 8) + i;
    const float v = x[(b * L_ + l) * NV + n];
    const float m = (v != 0.0f || n == 0) ? 1.0f : 0.0f;
    sv += m; sx += m * v; sxx += m * v * v;
  }
  __shared__ float ssv[8][32], ssx[8][32], ssxx[8][32];
  ssv[slice][threadIdx.x & 31] = sv;
  ssx[slice][threadIdx.x & 31] = sx;
  ssxx[slice][threadIdx.x & 31] = sxx;
  __syncthreads();
  if (slice == 0) {
    const int c = threadIdx.x & 31;
    float tv = 0.f, tx = 0.f, txx = 0.f;
#pragma unroll
    for (int r = 0; r < 8; ++r) { tv += ssv[r][c]; tx += ssx[r][c]; txx += ssxx[r][c]; }
    const float cnt = tv + 1e-8f;
    const float mean = tx / cnt;
    const float var = (txx - 2.0f * mean * tx + mean * mean * tv) / cnt;
    means[b * NV + n] = mean;
    stdev[b * NV + n] = sqrtf(var + 1e-8f);
  }
}

// build bf16 token matrix xc[B,LM,L]: normalized variates + calendar marks
__global__ void build_tokens_kernel(const float* __restrict__ x_enc,
                                    const float* __restrict__ x_mark,
                                    const float* __restrict__ means,
                                    const float* __restrict__ stdev,
                                    bf16* __restrict__ xc) {
  int idx = blockIdx.x * 256 + threadIdx.x;  // over B*L*LM, tok fastest
  if (idx >= B_ * L_ * LM) return;
  const int tok = idx % LM;
  const int t2 = idx / LM;
  const int l = t2 % L_;
  const int b = t2 / L_;
  float val;
  if (tok < NV) {
    const float v = x_enc[(b * L_ + l) * NV + tok];
    const float m = (v != 0.0f || tok == 0) ? 1.0f : 0.0f;
    val = (v - means[b * NV + tok]) / stdev[b * NV + tok] * m;
  } else {
    val = x_mark[(b * L_ + l) * NMARK + (tok - NV)];
  }
  xc[(b * LM + tok) * L_ + l] = (bf16)val;
}

// depthwise causal conv (k=2) + SiLU on x-half of xz; direction-aware
__global__ void mamba_pre_kernel(const float* __restrict__ xz,
                                 const float* __restrict__ conv_w,
                                 const float* __restrict__ conv_b,
                                 float* __restrict__ xact, bf16* __restrict__ xact_bf,
                                 int rev) {
  int idx = blockIdx.x * 256 + threadIdx.x;  // MTOK * DMODEL
  if (idx >= MTOK * DMODEL) return;
  const int d = idx % DMODEL;
  const int row = idx / DMODEL;
  const int t = row % LM;
  const float x0 = xz[row * (2 * DMODEL) + d];
  float xp = 0.0f;
  if (!rev) { if (t > 0)      xp = xz[(row - 1) * (2 * DMODEL) + d]; }
  else      { if (t < LM - 1) xp = xz[(row + 1) * (2 * DMODEL) + d]; }
  const float w0 = conv_w[d * 2 + 0], w1 = conv_w[d * 2 + 1];
  const float c = w0 * xp + w1 * x0 + conv_b[d];
  const float a = c / (1.0f + __expf(-c));
  xact[idx] = a;
  xact_bf[idx] = (bf16)a;
}

__global__ void split_dt_kernel(const float* __restrict__ xdbl, bf16* __restrict__ dtraw) {
  int idx = blockIdx.x * 256 + threadIdx.x;  // MTOK * DTR
  if (idx >= MTOK * DTR) return;
  const int r = idx % DTR;
  const int row = idx / DTR;
  dtraw[idx] = (bf16)xdbl[row * (DTR + 2 * DSTATE) + r];
}

// selective scan: one thread per (b,d), h[16] in registers; fuses softplus,
// D-skip and SiLU(z) gate; writes bf16 for out_proj GEMM.
__global__ __launch_bounds__(256)
void scan_kernel(const float* __restrict__ dtlin, const float* __restrict__ xact,
                 const float* __restrict__ xz, const float* __restrict__ xdbl,
                 const float* __restrict__ dt_b, const float* __restrict__ A_log,
                 const float* __restrict__ Dvec, bf16* __restrict__ ybf, int rev) {
  const int b = blockIdx.x >> 1;
  const int d = ((blockIdx.x & 1) << 8) + threadIdx.x;
  float Aa[DSTATE], h[DSTATE];
#pragma unroll
  for (int s = 0; s < DSTATE; ++s) {
    Aa[s] = -__expf(A_log[d * DSTATE + s]);
    h[s] = 0.0f;
  }
  const float db = dt_b[d];
  const float Dv = Dvec[d];
  for (int step = 0; step < LM; ++step) {
    const int t = rev ? (LM - 1 - step) : step;
    const int row = b * LM + t;
    float dt = dtlin[row * DMODEL + d] + db;
    dt = (dt > 20.0f) ? dt : log1pf(__expf(dt));
    const float u = xact[row * DMODEL + d];
    const float z = xz[row * (2 * DMODEL) + DMODEL + d];
    const float* BC = xdbl + row * (DTR + 2 * DSTATE);
    const float du = dt * u;
    float y = 0.0f;
#pragma unroll
    for (int s = 0; s < DSTATE; ++s) {
      const float dA = __expf(dt * Aa[s]);
      h[s] = dA * h[s] + du * BC[DTR + s];
      y = fmaf(h[s], BC[DTR + DSTATE + s], y);
    }
    y += u * Dv;
    y *= z / (1.0f + __expf(-z));
    ybf[row * DMODEL + d] = (bf16)y;
  }
}

// layernorm over DMODEL of (a [+ b]); writes f32 + bf16 mirror. One row/block.
__global__ __launch_bounds__(128)
void ln_kernel(const float* __restrict__ a, const float* __restrict__ bsum,
               const float* __restrict__ g, const float* __restrict__ bb,
               float* __restrict__ outf, bf16* __restrict__ outbf) {
  const int row = blockIdx.x;
  const int tid = threadIdx.x;
  const float* ra = a + row * DMODEL;
  const float* rb = bsum ? bsum + row * DMODEL : nullptr;
  float v[4];
  float s = 0.f, ss = 0.f;
#pragma unroll
  for (int i = 0; i < 4; ++i) {
    const int c = i * 128 + tid;
    float x = ra[c];
    if (rb) x += rb[c];
    v[i] = x; s += x; ss += x * x;
  }
  __shared__ float s1[128], s2[128];
  s1[tid] = s; s2[tid] = ss;
  __syncthreads();
  for (int off = 64; off; off >>= 1) {
    if (tid < off) { s1[tid] += s1[tid + off]; s2[tid] += s2[tid + off]; }
    __syncthreads();
  }
  __shared__ float mean_s, rstd_s;
  if (tid == 0) {
    const float mean = s1[0] / (float)DMODEL;
    const float var = s2[0] / (float)DMODEL - mean * mean;
    mean_s = mean; rstd_s = rsqrtf(var + 1e-5f);
  }
  __syncthreads();
  const float mean = mean_s, rstd = rstd_s;
#pragma unroll
  for (int i = 0; i < 4; ++i) {
    const int c = i * 128 + tid;
    const float o = (v[i] - mean) * rstd * g[c] + bb[c];
    outf[row * DMODEL + c] = o;
    if (outbf) outbf[row * DMODEL + c] = (bf16)o;
  }
}

__global__ void denorm_kernel(const float* __restrict__ dec,
                              const float* __restrict__ means,
                              const float* __restrict__ stdev,
                              float* __restrict__ out) {
  int idx = blockIdx.x * 256 + threadIdx.x;  // B*PLEN*NV, n fastest
  if (idx >= B_ * PLEN * NV) return;
  const int n = idx % NV;
  const int t2 = idx / NV;
  const int pl = t2 % PLEN;
  const int b = t2 / PLEN;
  out[idx] = dec[(b * LM + n) * PLEN + pl] * stdev[b * NV + n] + means[b * NV + n];
}

// ---------------------------------------------------------------------------
// Host orchestration
// ---------------------------------------------------------------------------
static void launch_gemm(hipStream_t s, const bf16* A, const bf16* W, const float* bias,
                        float* C, bf16* Cbf, int M, int N, int K,
                        int act, int acc, int nt) {
  if (nt == 4) {
    dim3 g(M / 32, N / 64);
    if (acc)           gemm_wmma_kernel<2, 4, 0, 1, 0><<<g, 32, 0, s>>>(A, W, bias, C, nullptr, M, N, K);
    else if (act == 1) gemm_wmma_kernel<2, 4, 1, 0, 1><<<g, 32, 0, s>>>(A, W, bias, C, Cbf, M, N, K);
    else if (Cbf)      gemm_wmma_kernel<2, 4, 0, 0, 1><<<g, 32, 0, s>>>(A, W, bias, C, Cbf, M, N, K);
    else               gemm_wmma_kernel<2, 4, 0, 0, 0><<<g, 32, 0, s>>>(A, W, bias, C, nullptr, M, N, K);
  } else {
    dim3 g(M / 32, N / 32);
    gemm_wmma_kernel<2, 2, 0, 0, 0><<<g, 32, 0, s>>>(A, W, bias, C, nullptr, M, N, K);
  }
}

extern "C" void kernel_launch(void* const* d_in, const int* in_sizes, int n_in,
                              void* d_out, int out_size, void* d_ws, size_t ws_size,
                              hipStream_t stream) {
  (void)in_sizes; (void)n_in; (void)out_size; (void)ws_size;
  auto in = [&](int i) { return (const float*)d_in[i]; };

  // workspace carve-out (256B aligned)
  char* wp = (char*)d_ws;
  auto carve = [&](size_t bytes) {
    char* p = wp;
    wp += (bytes + 255) & ~(size_t)255;
    return p;
  };
  float* means   = (float*)carve(B_ * NV * 4);
  float* stdev   = (float*)carve(B_ * NV * 4);
  bf16*  xc_bf   = (bf16*) carve((size_t)MTOK * L_ * 2);
  float* enc_f   = (float*)carve((size_t)MTOK * DMODEL * 4);
  bf16*  enc_bf  = (bf16*) carve((size_t)MTOK * DMODEL * 2);
  float* newx_f  = (float*)carve((size_t)MTOK * DMODEL * 4);
  float* xz_f    = (float*)carve((size_t)MTOK * 2 * DMODEL * 4);  // also FFN hidden f32
  float* xact_f  = (float*)carve((size_t)MTOK * DMODEL * 4);
  bf16*  xact_bf = (bf16*) carve((size_t)MTOK * DMODEL * 2);      // also FFN hidden bf16
  float* xdbl_f  = (float*)carve((size_t)MTOK * (DTR + 2 * DSTATE) * 4);
  bf16*  dtraw_bf= (bf16*) carve((size_t)MTOK * DTR * 2);
  float* dtlin_f = (float*)carve((size_t)MTOK * DMODEL * 4);
  bf16*  yscan_bf= (bf16*) carve((size_t)MTOK * DMODEL * 2);
  bf16*  wbuf    = (bf16*) carve((size_t)2195456 * 2);            // per-layer bf16 weights
  float* dec_f   = (float*)carve((size_t)MTOK * PLEN * 4);

  // per-layer weight slots inside wbuf (element offsets)
  bf16* w_ip[2]  = { wbuf,            wbuf + 524288 };            // in_proj  [1024,512]
  bf16* w_xp[2]  = { wbuf + 1048576,  wbuf + 1081344 };           // x_proj   [64,512]
  bf16* w_dt[2]  = { wbuf + 1114112,  wbuf + 1130496 };           // dt_w     [512,32]
  bf16* w_op[2]  = { wbuf + 1146880,  wbuf + 1409024 };           // out_proj [512,512]
  bf16* w_c1     = wbuf + 1671168;                                // conv1    [512,512]
  bf16* w_c2     = wbuf + 1933312;                                // conv2    [512,512]

  auto cvt = [&](const float* s, bf16* d, int n) {
    cvt_bf16_kernel<<<(n + 255) / 256, 256, 0, stream>>>(s, d, n);
  };

  // 1) masked stats + token build
  stats_kernel<<<dim3(B_, NV / 32), 256, 0, stream>>>(in(0), means, stdev);
  build_tokens_kernel<<<(B_ * L_ * LM + 255) / 256, 256, 0, stream>>>(
      in(0), in(1), means, stdev, xc_bf);

  // 2) inverted embedding: enc = xc @ emb_w^T + emb_b
  cvt(in(2), w_ip[0], DMODEL * L_);
  launch_gemm(stream, xc_bf, w_ip[0], in(3), enc_f, enc_bf, MTOK, DMODEL, L_, 0, 0, 4);

  // 3) encoder layers
  for (int li = 0; li < 3; ++li) {
    const int base = 4 + li * 26;
    // mamba param index offsets within layer: fwd at +0, rev at +9
    for (int r = 0; r < 2; ++r) {
      const int mb = base + r * 9;
      cvt(in(mb + 0), w_ip[r], 2 * DMODEL * DMODEL);
      cvt(in(mb + 3), w_xp[r], (DTR + 2 * DSTATE) * DMODEL);
      cvt(in(mb + 4), w_dt[r], DMODEL * DTR);
      cvt(in(mb + 8), w_op[r], DMODEL * DMODEL);
    }
    cvt(in(base + 18), w_c1, DFF * DMODEL);
    cvt(in(base + 20), w_c2, DMODEL * DFF);

    for (int rev = 0; rev < 2; ++rev) {
      const int mb = base + rev * 9;
      // xz = enc @ in_proj^T  [MTOK, 1024]
      launch_gemm(stream, enc_bf, w_ip[rev], nullptr, xz_f, nullptr,
                  MTOK, 2 * DMODEL, DMODEL, 0, 0, 4);
      // causal depthwise conv + SiLU
      mamba_pre_kernel<<<(MTOK * DMODEL + 255) / 256, 256, 0, stream>>>(
          xz_f, in(mb + 1), in(mb + 2), xact_f, xact_bf, rev);
      // xdbl = x @ x_proj^T  [MTOK, 64]
      launch_gemm(stream, xact_bf, w_xp[rev], nullptr, xdbl_f, nullptr,
                  MTOK, DTR + 2 * DSTATE, DMODEL, 0, 0, 4);
      split_dt_kernel<<<(MTOK * DTR + 255) / 256, 256, 0, stream>>>(xdbl_f, dtraw_bf);
      // dt_lin = dt_raw @ dt_w^T  [MTOK, 512]
      launch_gemm(stream, dtraw_bf, w_dt[rev], nullptr, dtlin_f, nullptr,
                  MTOK, DMODEL, DTR, 0, 0, 4);
      // selective scan (+softplus, D-skip, SiLU gate)
      scan_kernel<<<B_ * 2, 256, 0, stream>>>(
          dtlin_f, xact_f, xz_f, xdbl_f, in(mb + 5), in(mb + 6), in(mb + 7),
          yscan_bf, rev);
      // new_x (+)= y @ out_proj^T
      launch_gemm(stream, yscan_bf, w_op[rev], nullptr, newx_f, nullptr,
                  MTOK, DMODEL, DMODEL, 0, rev ? 1 : 0, 4);
    }
    // x = LN(enc + new_x)
    ln_kernel<<<MTOK, 128, 0, stream>>>(enc_f, newx_f, in(base + 22), in(base + 23),
                                        enc_f, enc_bf);
    // FFN: hidden = relu(x @ conv1^T + b1)   (hidden reuses xz_f / xact_bf regions)
    float* hid_f = xz_f;
    bf16*  hid_bf = xact_bf;
    launch_gemm(stream, enc_bf, w_c1, in(base + 19), hid_f, hid_bf,
                MTOK, DFF, DMODEL, 1, 0, 4);
    launch_gemm(stream, hid_bf, w_c2, in(base + 21), newx_f, nullptr,
                MTOK, DMODEL, DFF, 0, 0, 4);
    // x = LN(x + ffn)
    ln_kernel<<<MTOK, 128, 0, stream>>>(enc_f, newx_f, in(base + 24), in(base + 25),
                                        enc_f, enc_bf);
  }

  // 4) final norm + projection + de-normalization
  ln_kernel<<<MTOK, 128, 0, stream>>>(enc_f, nullptr, in(82), in(83), enc_f, enc_bf);
  cvt(in(84), w_ip[0], PLEN * DMODEL);
  launch_gemm(stream, enc_bf, w_ip[0], in(85), dec_f, nullptr, MTOK, PLEN, DMODEL, 0, 0, 2);
  denorm_kernel<<<(B_ * PLEN * NV + 255) / 256, 256, 0, stream>>>(
      dec_f, means, stdev, (float*)d_out);
}
